// DotProductAttention_36283883716848
// MI455X (gfx1250) — matrix operand order repaired and checked by reading, not compiled, not run on hardware
//
#include <hip/hip_runtime.h>
#include <stdint.h>

// CDNA5 / gfx1250 fused attention:
//   score = Q @ V^T  (bf16 WMMA, f32 accum)
//   attn  = softmax(score) (f32, in 320KB LDS) -> async LDS->global DMA store
//   ctx   = attn @ V (bf16 WMMA, f32 accum)
// Outputs (flat, concatenated): context [B,Lq,H] then attn [B,Lq,Lv], both f32.

typedef __attribute__((ext_vector_type(16))) __bf16 v16bf;
typedef __attribute__((ext_vector_type(8)))  float  v8f;

constexpr int Bn = 8, Lq = 2048, Lv = 2048, Hd = 256;
constexpr int TQ = 32;        // query rows per workgroup
constexpr int TV = 64;        // value rows per staged chunk
constexpr int THREADS = 256;  // 8 wave32
constexpr int VPRELOAD = TV * Hd / 4 / THREADS;  // float4 per thread per chunk = 16

__device__ __forceinline__ unsigned short f32_to_bf16(float f) {
  union { float f; unsigned int u; } c; c.f = f;
  unsigned int u = c.u;
  unsigned int r = (u + 0x7FFFu + ((u >> 16) & 1u)) >> 16;  // RNE
  return (unsigned short)r;
}

union FragBF { uint4 q[2]; v16bf v; unsigned short u[16]; };

__global__ __launch_bounds__(THREADS, 1)
void attn_fused_kernel(const float* __restrict__ Q, const float* __restrict__ V,
                       float* __restrict__ ctx_out, float* __restrict__ attn_out)
{
  __shared__ float          sS[TQ * Lv];      // 256 KB: scores -> probs
  __shared__ unsigned short sQ[TQ * Hd];      //  16 KB: Q tile, bf16 row-major
  __shared__ unsigned short sV[TV * Hd];      //  32 KB: V chunk
  __shared__ float          sRed[TQ][8];      //   1 KB: per-row partials

  const int wg   = blockIdx.x;
  const int b    = wg / (Lq / TQ);
  const int q0   = (wg % (Lq / TQ)) * TQ;
  const int tid  = threadIdx.x;
  const int lane = tid & 31;
  const int wave = tid >> 5;                  // 0..7

  const float* Qb = Q + (size_t)b * Lq * Hd;
  const float* Vb = V + (size_t)b * Lv * Hd;
  const float4* v4 = (const float4*)Vb;

  // ---------------- stage Q tile (32x256 f32 -> bf16 LDS) ----------------
  {
    const float4* q4 = (const float4*)(Qb + (size_t)q0 * Hd);
    for (int i = tid; i < TQ * Hd / 4; i += THREADS) {
      float4 v = q4[i];
      unsigned short* dst = &sQ[i * 4];
      dst[0] = f32_to_bf16(v.x); dst[1] = f32_to_bf16(v.y);
      dst[2] = f32_to_bf16(v.z); dst[3] = f32_to_bf16(v.w);
    }
  }

  // software pipeline: preload V chunk 0 into registers (overlaps barrier)
  float4 pre[VPRELOAD];
  #pragma unroll
  for (int j = 0; j < VPRELOAD; ++j) pre[j] = v4[tid + j * THREADS];
  __syncthreads();

  // ---------------- phase 1: S = Q @ V^T into LDS ----------------
  const int rt = wave & 1;   // row tile (16 q rows)
  const int ct = wave >> 1;  // col tile 0..3 within 64-col chunk
  for (int vc = 0; vc < Lv / TV; ++vc) {
    // convert preloaded regs -> bf16 LDS (row-major)
    #pragma unroll
    for (int j = 0; j < VPRELOAD; ++j) {
      const int i = tid + j * THREADS;
      unsigned short* dst = &sV[i * 4];
      float4 v = pre[j];
      dst[0] = f32_to_bf16(v.x); dst[1] = f32_to_bf16(v.y);
      dst[2] = f32_to_bf16(v.z); dst[3] = f32_to_bf16(v.w);
    }
    __syncthreads();

    // issue next chunk's global loads now; first use is next iteration,
    // so the latency hides behind the WMMA block below.
    if (vc + 1 < Lv / TV) {
      #pragma unroll
      for (int j = 0; j < VPRELOAD; ++j)
        pre[j] = v4[(vc + 1) * (TV * Hd / 4) + tid + j * THREADS];
    }
    if (vc + 2 < Lv / TV)
      __builtin_prefetch(Vb + (size_t)(vc + 2) * TV * Hd + tid * 16, 0, 0);

    v8f acc;
    #pragma unroll
    for (int r = 0; r < 8; ++r) acc[r] = 0.f;

    #pragma unroll
    for (int k0 = 0; k0 < Hd; k0 += 32) {
      // A fragment: Q rows 16x32 bf16 (lane<16: K=e / 16+e; lane>=16: +8)
      FragBF A;
      {
        const int row = rt * 16 + (lane & 15);
        const int kb  = k0 + ((lane >> 4) << 3);
        A.q[0] = *(const uint4*)&sQ[row * Hd + kb];
        A.q[1] = *(const uint4*)&sQ[row * Hd + kb + 16];
      }
      // B fragment: (V^T) 32x16: N = lane&15 -> V row, K = e / 16+e (lane>=16: +8)
      FragBF Bf;
      {
        const int vrow = ct * 16 + (lane & 15);
        const int kb   = k0 + ((lane >> 4) << 3);
        Bf.q[0] = *(const uint4*)&sV[vrow * Hd + kb];
        Bf.q[1] = *(const uint4*)&sV[vrow * Hd + kb + 16];
      }
      acc = __builtin_amdgcn_wmma_f32_16x16x32_bf16(false, A.v, false, Bf.v,
                                                    (short)0, acc, false, false);
    }
    // scatter D into sS (C/D: M = r + (lane>=16)*8, N = lane&15)
    {
      const int col   = vc * TV + ct * 16 + (lane & 15);
      const int rbase = rt * 16 + ((lane >> 4) << 3);
      #pragma unroll
      for (int r = 0; r < 8; ++r)
        sS[(rbase + r) * Lv + col] = acc[r];
    }
    __syncthreads();
  }

  // ---------------- phase 1b: row softmax in LDS ----------------
  const int srow  = tid >> 3;            // 0..31
  const int spart = tid & 7;             // 0..7
  const int sbase = spart * (Lv / 8);    // 256-elem segment
  float* srowp = &sS[srow * Lv + sbase];

  float m = -3.402823466e+38f;
  for (int i = 0; i < Lv / 8; ++i) m = fmaxf(m, srowp[i]);
  sRed[srow][spart] = m;
  __syncthreads();
  float rowm = sRed[srow][0];
  #pragma unroll
  for (int i = 1; i < 8; ++i) rowm = fmaxf(rowm, sRed[srow][i]);
  __syncthreads();

  float s = 0.f;
  for (int i = 0; i < Lv / 8; ++i) {
    float e = __expf(srowp[i] - rowm);
    srowp[i] = e;
    s += e;
  }
  sRed[srow][spart] = s;
  __syncthreads();
  float rows = sRed[srow][0];
  #pragma unroll
  for (int i = 1; i < 8; ++i) rows += sRed[srow][i];
  const float inv = 1.0f / rows;

  for (int i = 0; i < Lv / 8; ++i) srowp[i] *= inv;   // normalize in LDS
  __syncthreads();

  // ---------------- async DMA: attn tile LDS -> global ----------------
  // This WG's attn region is one contiguous 32*2048*4 = 256KB block.
  // GLOBAL_STORE_ASYNC_FROM_LDS_B128 (ASYNCcnt) streams it out while the
  // second GEMM below runs; S_ENDPGM's implicit wait-idle covers completion.
  {
    float* attn_base = attn_out + ((size_t)b * Lq + q0) * (size_t)Lv;
    for (int i = tid; i < TQ * Lv / 4; i += THREADS) {
      unsigned lds_off = (unsigned)(uintptr_t)(&sS[i * 4]);   // LDS aperture: low 32 bits
      unsigned long long ga = (unsigned long long)(uintptr_t)(attn_base + i * 4);
      asm volatile("global_store_async_from_lds_b128 %0, %1, off"
                   :: "v"(ga), "v"(lds_off)
                   : "memory");
    }
  }

  // ---------------- phase 2: ctx = P @ V ----------------
  v8f accum[4];
  #pragma unroll
  for (int t = 0; t < 4; ++t)
    #pragma unroll
    for (int r = 0; r < 8; ++r) accum[t][r] = 0.f;

  // preload chunk 0 (registers only; sV untouched until after next barrier)
  #pragma unroll
  for (int j = 0; j < VPRELOAD; ++j) pre[j] = v4[tid + j * THREADS];

  for (int vc = 0; vc < Lv / TV; ++vc) {
    __syncthreads();  // prior sV readers done before overwrite
    // stage V chunk TRANSPOSED: sV[h * TV + vloc]
    #pragma unroll
    for (int j = 0; j < VPRELOAD; ++j) {
      const int i    = tid + j * THREADS;
      const int vloc = i >> 6;          // i / (Hd/4)
      const int h4   = (i & 63) << 2;   // (i % (Hd/4)) * 4
      float4 val = pre[j];
      sV[(h4 + 0) * TV + vloc] = f32_to_bf16(val.x);
      sV[(h4 + 1) * TV + vloc] = f32_to_bf16(val.y);
      sV[(h4 + 2) * TV + vloc] = f32_to_bf16(val.z);
      sV[(h4 + 3) * TV + vloc] = f32_to_bf16(val.w);
    }
    __syncthreads();

    if (vc + 1 < Lv / TV) {
      #pragma unroll
      for (int j = 0; j < VPRELOAD; ++j)
        pre[j] = v4[(vc + 1) * (TV * Hd / 4) + tid + j * THREADS];
    }

    #pragma unroll
    for (int t = 0; t < 4; ++t) {
      const int ct2 = (wave >> 1) * 4 + t;
      #pragma unroll
      for (int kk = 0; kk < TV; kk += 32) {
        // A: P rows 16x32, f32 -> bf16 on the fly (same A layout)
        FragBF A;
        {
          const int row = rt * 16 + (lane & 15);
          const int kb  = kk + ((lane >> 4) << 3);
          const float* src = &sS[row * Lv + vc * TV + kb];
          #pragma unroll
          for (int e = 0; e < 8; ++e) A.u[e]     = f32_to_bf16(src[e]);
          #pragma unroll
          for (int e = 0; e < 8; ++e) A.u[8 + e] = f32_to_bf16(src[16 + e]);
        }
        // B: V 32x16 from transposed staging: N = lane&15 (h), K = e / 16+e (lane>=16: +8)
        FragBF Bf;
        {
          const int hcol = ct2 * 16 + (lane & 15);
          const int kb2  = kk + ((lane >> 4) << 3);
          Bf.q[0] = *(const uint4*)&sV[hcol * TV + kb2];
          Bf.q[1] = *(const uint4*)&sV[hcol * TV + kb2 + 16];
        }
        accum[t] = __builtin_amdgcn_wmma_f32_16x16x32_bf16(false, A.v, false, Bf.v,
                                                           (short)0, accum[t], false, false);
      }
    }
  }

  // write context
  {
    const int rbase = rt * 16 + ((lane >> 4) << 3);
    #pragma unroll
    for (int t = 0; t < 4; ++t) {
      const int col = ((wave >> 1) * 4 + t) * 16 + (lane & 15);
      #pragma unroll
      for (int r = 0; r < 8; ++r)
        ctx_out[((size_t)b * Lq + q0 + rbase + r) * Hd + col] = accum[t][r];
    }
  }
}

extern "C" void kernel_launch(void* const* d_in, const int* in_sizes, int n_in,
                              void* d_out, int out_size, void* d_ws, size_t ws_size,
                              hipStream_t stream) {
  const float* Q = (const float*)d_in[0];
  const float* V = (const float*)d_in[1];
  float* ctx  = (float*)d_out;                          // [B,Lq,H]
  float* attn = (float*)d_out + (size_t)Bn * Lq * Hd;   // [B,Lq,Lv]
  dim3 grid(Bn * (Lq / TQ));   // 512 workgroups
  attn_fused_kernel<<<grid, THREADS, 0, stream>>>(Q, V, ctx, attn);
}
